// MoE_32238024524134
// MI455X (gfx1250) — matrix operand order repaired and checked by reading, not compiled, not run on hardware
//
#include <hip/hip_runtime.h>
#include <hip/hip_bf16.h>

// ---------- types for WMMA ----------
typedef __attribute__((ext_vector_type(16))) __bf16 v16bf;
typedef __attribute__((ext_vector_type(8)))  float  v8f;
typedef __attribute__((ext_vector_type(4)))  unsigned int v4u;

union Frag16 {            // 32 bytes: 16 bf16 elements = 8 VGPRs
    v16bf v;
    v4u   q[2];
};

// ---------- helpers ----------
__device__ __forceinline__ unsigned short f2bf(float f) {
    unsigned u = __float_as_uint(f);
    u = u + 0x7FFFu + ((u >> 16) & 1u);   // round-to-nearest-even
    return (unsigned short)(u >> 16);
}
__device__ __forceinline__ float bf2f(unsigned short h) {
    return __uint_as_float(((unsigned)h) << 16);
}

// ---------- conversion kernels ----------
__global__ __launch_bounds__(256) void f32_to_bf16_kernel(
    const float* __restrict__ in, unsigned short* __restrict__ out, long n)
{
    long i = (long)blockIdx.x * blockDim.x + threadIdx.x;
    if (i < n) out[i] = f2bf(in[i]);
}

// in: [K, N] f32 (row-major), out: [N, K] bf16 (row-major) -> B^T for GEMM
__global__ __launch_bounds__(256) void transpose_f32_to_bf16_kernel(
    const float* __restrict__ in, unsigned short* __restrict__ out,
    int K, int N, long inBatchStride, long outBatchStride)
{
    __shared__ float tile[32][33];
    const float*    src = in  + (long)blockIdx.z * inBatchStride;
    unsigned short* dst = out + (long)blockIdx.z * outBatchStride;
    int n0 = blockIdx.x * 32;
    int k0 = blockIdx.y * 32;
    int tx = threadIdx.x;            // 0..31
    int ty = threadIdx.y;            // 0..7
    #pragma unroll
    for (int i = 0; i < 32; i += 8)
        tile[ty + i][tx] = src[(long)(k0 + ty + i) * N + (n0 + tx)];
    __syncthreads();
    #pragma unroll
    for (int i = 0; i < 32; i += 8)
        dst[(long)(n0 + ty + i) * K + (k0 + tx)] = f2bf(tile[tx][ty + i]);
}

// ---------- main bf16 WMMA GEMM with async global->LDS double buffering ----
// C[M,N] = op(A[M,K] * B[K,N] + bias[N]),  B supplied transposed: Bt[N,K]
// Block: 256 threads = 8 waves (wave32). Block tile 128(M) x 64*NSUB(N),
// K-chunk 64. Slabs staged via global_load_async_to_lds_b128 (ASYNCcnt),
// double-buffered; fragments come from LDS (144B padded rows -> conflict-free
// ds_load_b128). Wave grid 2(M) x 4(N); wave tile 64(M) x 16*NSUB(N).
#define GEMM_ROWB 144                       // padded bytes per 64-element row
#define GEMM_ABUF (128 * GEMM_ROWB)         // 18432 bytes per A slab buffer

template<int NSUB, bool RELU, bool OUT_BF16>
__global__ __launch_bounds__(256) void gemm_bf16_wmma_kernel(
    const unsigned short* __restrict__ A,  int lda,
    const unsigned short* __restrict__ Bt, int ldb,
    const float* __restrict__ bias,
    void* __restrict__ Cout, int ldc,
    int K)
{
    extern __shared__ char smem[];          // A0|A1|B0|B1 (dynamic, offset 0)
    constexpr unsigned ABUF = GEMM_ABUF;
    constexpr unsigned BBUF = (unsigned)(64 * NSUB) * GEMM_ROWB;
    constexpr int      BN   = 64 * NSUB;    // block N tile

    const int tid  = threadIdx.x;
    const int wave = tid >> 5;
    const int lane = tid & 31;
    const int l    = lane & 15;
    const int hi   = lane >> 4;             // half-wave select

    const int wm = wave & 1;                // 0..1
    const int wn = wave >> 1;               // 0..3
    const int m0 = blockIdx.y * 128 + wm * 64;
    const int n0 = blockIdx.x * BN + wn * (16 * NSUB);

    // ---- async slab loader: every wave loads part of both A and B slabs ---
    // Slab rows are 64 elems (128B) split into 64B half-rows; one async b128
    // issue moves 8 half-rows (4 lanes per half-row, 16B per lane).
    const int      sub   = lane >> 2;                  // 0..7
    const int      colB  = (lane & 3) * 16;            // byte in 64B half-row
    const unsigned ldbyA = (unsigned)lda * 2;
    const unsigned ldbyB = (unsigned)ldb * 2;
    // half-row hr = wave*(8*CNT) + t*8 + sub  ->  row = base + t*4, half=sub&1
    const unsigned goffA0 = (unsigned)((wave * 16 + (sub >> 1)) * ldbyA
                                       + (sub & 1) * 64 + colB);
    const unsigned goffB0 = (unsigned)((wave * (NSUB * 8) + (sub >> 1)) * ldbyB
                                       + (sub & 1) * 64 + colB);
    const unsigned loffA0 = (unsigned)((wave * 16 + (sub >> 1)) * GEMM_ROWB
                                       + (sub & 1) * 64 + colB);
    const unsigned loffB0 = 2u * ABUF
                          + (unsigned)((wave * (NSUB * 8) + (sub >> 1)) * GEMM_ROWB
                                       + (sub & 1) * 64 + colB);
    const unsigned long long aorig = (unsigned long long)(const void*)
        (A + (size_t)blockIdx.y * 128 * lda);
    const unsigned long long borig = (unsigned long long)(const void*)
        (Bt + (size_t)blockIdx.x * BN * ldb);

    auto issue_slabs = [&](unsigned long long kb, int parity) {
        const unsigned long long saA = aorig + kb;    // block-uniform SADDR
        const unsigned long long saB = borig + kb;
        const unsigned pa = parity ? ABUF : 0u;
        const unsigned pb = parity ? BBUF : 0u;
        #pragma unroll
        for (int t = 0; t < 4; ++t) {                 // A slab share
            unsigned va = goffA0 + (unsigned)(t * 4) * ldbyA;
            unsigned ld = loffA0 + pa + (unsigned)(t * 4 * GEMM_ROWB);
            asm volatile("global_load_async_to_lds_b128 %0, %1, %2"
                         :: "v"(ld), "v"(va), "s"(saA) : "memory");
        }
        #pragma unroll
        for (int t = 0; t < 2 * NSUB; ++t) {          // B slab share
            unsigned va = goffB0 + (unsigned)(t * 4) * ldbyB;
            unsigned ld = loffB0 + pb + (unsigned)(t * 4 * GEMM_ROWB);
            asm volatile("global_load_async_to_lds_b128 %0, %1, %2"
                         :: "v"(ld), "v"(va), "s"(saB) : "memory");
        }
    };

    v8f acc[4][NSUB];
    #pragma unroll
    for (int i = 0; i < 4; ++i)
        #pragma unroll
        for (int j = 0; j < NSUB; ++j)
            acc[i][j] = (v8f){0.f,0.f,0.f,0.f,0.f,0.f,0.f,0.f};

    const int nsteps = K >> 6;              // K / 64

    issue_slabs(0ull, 0);                   // prefetch first slab pair

    for (int s = 0; s < nsteps; ++s) {
        asm volatile("s_wait_asynccnt 0x0" ::: "memory");
        __syncthreads();                    // slab(parity s) visible to all

        if (s + 1 < nsteps)
            issue_slabs((unsigned long long)(s + 1) << 7, (s + 1) & 1);

        const char* ab = smem + ((s & 1) ? ABUF : 0u);
        const char* bb = smem + 2u * ABUF + ((s & 1) ? BBUF : 0u);

        #pragma unroll
        for (int kh = 0; kh < 2; ++kh) {    // two 32-wide K-steps per slab
            Frag16 b[NSUB];
            // B operand (32x16), contiguous k order used here: lane<16 -> K = k0+[0..15]
            #pragma unroll
            for (int j = 0; j < NSUB; ++j) {
                const char* q = bb + (wn * (16 * NSUB) + j * 16 + l) * GEMM_ROWB
                                   + kh * 64 + hi * 32;
                b[j].q[0] = *(const v4u*)(q);
                b[j].q[1] = *(const v4u*)(q + 16);
            }
            // A operand (16x32), same contiguous k order as B: lane<16 -> K = k0+[0..15]
            #pragma unroll
            for (int i = 0; i < 4; ++i) {
                Frag16 a;
                const char* q = ab + (wm * 64 + i * 16 + l) * GEMM_ROWB
                                   + kh * 64 + hi * 32;
                a.q[0] = *(const v4u*)(q);
                a.q[1] = *(const v4u*)(q + 16);
                #pragma unroll
                for (int j = 0; j < NSUB; ++j)
                    acc[i][j] = __builtin_amdgcn_wmma_f32_16x16x32_bf16(
                        false, a.v, false, b[j].v,
                        (short)0, acc[i][j], false, false);
            }
        }
    }

    // Epilogue. C/D layout: VGPR r -> M = hi*8 + r, N = lane&15 (per tile).
    #pragma unroll
    for (int i = 0; i < 4; ++i) {
        #pragma unroll
        for (int j = 0; j < NSUB; ++j) {
            const int n  = n0 + j * 16 + l;
            const float bv = bias[n];
            #pragma unroll
            for (int r = 0; r < 8; ++r) {
                const int m = m0 + i * 16 + hi * 8 + r;
                float v = acc[i][j][r] + bv;
                if (RELU) v = fmaxf(v, 0.0f);
                if (OUT_BF16)
                    ((unsigned short*)Cout)[(long)m * ldc + n] = f2bf(v);
                else
                    ((float*)Cout)[(long)m * ldc + n] = v;
            }
        }
    }
}

// ---------- gate second layer + softmax over E=8 ----------
__global__ __launch_bounds__(256) void gate2_softmax_kernel(
    const unsigned short* __restrict__ g, const float* __restrict__ w2,
    const float* __restrict__ b2, float* __restrict__ scores, int D)
{
    const long t   = (long)blockIdx.x * blockDim.x + threadIdx.x;
    const int  e   = (int)(t & 7);
    const long row = t >> 3;

    const unsigned short* gr = g + row * D;
    float acc = b2[e];
    for (int d = 0; d < D; d += 2) {
        acc += bf2f(gr[d])     * w2[(long)d * 8 + e];
        acc += bf2f(gr[d + 1]) * w2[(long)(d + 1) * 8 + e];
    }
    float m = acc;
    #pragma unroll
    for (int o = 1; o < 8; o <<= 1) m = fmaxf(m, __shfl_xor(m, o, 8));
    float p = __expf(acc - m);
    float s = p;
    #pragma unroll
    for (int o = 1; o < 8; o <<= 1) s += __shfl_xor(s, o, 8);
    scores[row * 8 + e] = p / s;
}

// ---------- launcher ----------
extern "C" void kernel_launch(void* const* d_in, const int* in_sizes, int n_in,
                              void* d_out, int out_size, void* d_ws, size_t ws_size,
                              hipStream_t stream) {
    (void)in_sizes; (void)n_in; (void)out_size; (void)ws_size;

    const int D = 1024, H = 2048, E = 8, C = 1024, B = 8192;
    const unsigned LDS2 = 2u * GEMM_ABUF + 2u * 128u * GEMM_ROWB;  //  73728
    const unsigned LDS4 = 2u * GEMM_ABUF + 2u * 256u * GEMM_ROWB;  // 110592

    const float* x    = (const float*)d_in[0];
    const float* gw1  = (const float*)d_in[1];
    const float* gb1  = (const float*)d_in[2];
    const float* gw2  = (const float*)d_in[3];
    const float* gb2  = (const float*)d_in[4];
    const float* W1   = (const float*)d_in[5];
    const float* b1   = (const float*)d_in[6];
    const float* W2   = (const float*)d_in[7];
    const float* b2   = (const float*)d_in[8];
    const float* clfw = (const float*)d_in[9];
    const float* clfb = (const float*)d_in[10];
    float* out = (float*)d_out;

    // workspace layout (bytes)
    char* ws = (char*)d_ws;
    unsigned short* xb    = (unsigned short*)(ws + ((size_t)0   << 20)); // 16 MB [B,D]
    unsigned short* gw1t  = (unsigned short*)(ws + ((size_t)16  << 20)); //  2 MB [D,D]^T
    unsigned short* W1t   = (unsigned short*)(ws + ((size_t)18  << 20)); // 32 MB [E,H,D]
    unsigned short* W2t   = (unsigned short*)(ws + ((size_t)50  << 20)); // 64 MB [E,H,H]
    unsigned short* clfwt = (unsigned short*)(ws + ((size_t)114 << 20)); //  4 MB [C,H]
    unsigned short* gbuf  = (unsigned short*)(ws + ((size_t)118 << 20)); // 16 MB [B,D]
    unsigned short* hbuf  = (unsigned short*)(ws + ((size_t)134 << 20)); // 32 MB [B,H]
    unsigned short* ubuf  = (unsigned short*)(ws + ((size_t)166 << 20)); // 32 MB [B,H]

    // 1) convert activations / weights to bf16 (weights transposed to [N,K])
    {
        long n = (long)B * D;
        f32_to_bf16_kernel<<<(unsigned)((n + 255) / 256), 256, 0, stream>>>(x, xb, n);
    }
    dim3 tb(32, 8);
    transpose_f32_to_bf16_kernel<<<dim3(D / 32, D / 32, 1), tb, 0, stream>>>(
        gw1, gw1t, D, D, 0, 0);
    transpose_f32_to_bf16_kernel<<<dim3(H / 32, D / 32, E), tb, 0, stream>>>(
        W1, W1t, D, H, (long)D * H, (long)H * D);
    transpose_f32_to_bf16_kernel<<<dim3(H / 32, H / 32, E), tb, 0, stream>>>(
        W2, W2t, H, H, (long)H * H, (long)H * H);
    transpose_f32_to_bf16_kernel<<<dim3(C / 32, H / 32, 1), tb, 0, stream>>>(
        clfw, clfwt, H, C, 0, 0);

    // 2) router: g = relu(x @ gw1 + gb1), then scores = softmax(g @ gw2 + gb2)
    gemm_bf16_wmma_kernel<2, true, true>
        <<<dim3(D / 128, B / 128), 256, LDS2, stream>>>(
        xb, D, gw1t, D, gb1, gbuf, D, D);
    gate2_softmax_kernel<<<(B * E) / 256, 256, 0, stream>>>(
        gbuf, gw2, gb2, out + (size_t)B * E * C, D);

    // 3) experts: h = relu(x@W1[e]+b1[e]); u = h@W2[e]+b2[e]; logits_e = u@clfw+clfb
    for (int e = 0; e < E; ++e) {
        gemm_bf16_wmma_kernel<4, true, true>
            <<<dim3(H / 256, B / 128), 256, LDS4, stream>>>(
            xb, D, W1t + (size_t)e * H * D, D, b1 + (size_t)e * H, hbuf, H, D);
        gemm_bf16_wmma_kernel<4, false, true>
            <<<dim3(H / 256, B / 128), 256, LDS4, stream>>>(
            hbuf, H, W2t + (size_t)e * H * H, H, b2 + (size_t)e * H, ubuf, H, H);
        gemm_bf16_wmma_kernel<2, false, false>
            <<<dim3(C / 128, B / 128), 256, LDS2, stream>>>(
            ubuf, H, clfwt, H, clfb, out + (size_t)e * C, E * C, H);
    }
}
